// AdaptFormer_74543452389798
// MI455X (gfx1250) — matrix operand order repaired and checked by reading, not compiled, not run on hardware
//
#include <hip/hip_runtime.h>

#define DEPTH 4
#define CDIM 768
#define NHEAD 12
#define HDIM 64
#define FFDIM 3072
#define BOTD 64
#define NTOK 1568
#define BATCH 2
#define MROWS (BATCH * NTOK) /* 3136 */
#define KCONV 1536           /* 3*2*16*16 */

#define LDS_ROW 144          /* 64 f16 cols * 2B + 16B pad: spreads LDS banks */
#define LDS_SLAB (32 * LDS_ROW)

typedef __attribute__((ext_vector_type(16))) _Float16 v16h;
typedef __attribute__((ext_vector_type(8))) float v8f;

#define WMMA16(a, b, c) \
  __builtin_amdgcn_wmma_f32_16x16x32_f16(false, (a), false, (b), (short)0, (c), false, false)
// pin issue order: loads above, WMMAs below (avoids per-WMMA dscnt-0 waits)
#define SCHED_FENCE() __builtin_amdgcn_sched_barrier(0)

// ---------------- WMMA fragment loaders (per CDNA5 ISA 7.12.2 layouts) ----------
// A (16x32 f16, M x K): lane holds row (lane&15); K set = {8h..8h+7} U {16+8h..+7}
__device__ __forceinline__ v16h ld_a_frag(const _Float16* A, int lda, int m0, int k0, int lane) {
  int r = lane & 15, h = lane >> 4;
  const _Float16* p = A + (size_t)(m0 + r) * lda + k0 + 8 * h;
  union { v16h v; uint4 q[2]; } u;
  u.q[0] = *(const uint4*)(p);
  u.q[1] = *(const uint4*)(p + 16);
  return u.v;
}
// B (32x16 f16, K x N): lane holds column n0+(lane&15); K set = {8h..8h+7} U {16+8h..+7}
__device__ __forceinline__ v16h ld_b_frag(const _Float16* B, int ldb, int k0, int n0, int lane) {
  const _Float16* p = B + (size_t)(k0 + 8 * (lane >> 4)) * ldb + n0 + (lane & 15);
  v16h v;
#pragma unroll
  for (int s = 0; s < 8; ++s) {
    v[s] = p[(size_t)s * ldb];
    v[s + 8] = p[(size_t)(16 + s) * ldb];
  }
  return v;
}
// B fragment out of the LDS slab (padded rows)
__device__ __forceinline__ v16h ld_b_lds(const char* slab, int lane, int j) {
  const _Float16* p = (const _Float16*)(slab + (lane >> 4) * 8 * LDS_ROW + j * 32) + (lane & 15);
  v16h v;
#pragma unroll
  for (int s = 0; s < 8; ++s) {
    v[s] = p[s * (LDS_ROW / 2)];
    v[s + 8] = p[(16 + s) * (LDS_ROW / 2)];
  }
  return v;
}

// async copy of one 32xK-row x 64-col f16 B slab into LDS (128 threads, 2 b128 each)
__device__ __forceinline__ void slab_fill_async(const _Float16* Bg, int ldb, int n0, int k0,
                                                unsigned lds_base, int tid) {
#pragma unroll
  for (int c = tid; c < 256; c += 128) {
    int row = c >> 3, sub = c & 7;
    const _Float16* g = Bg + (size_t)(k0 + row) * ldb + n0 + sub * 8;
    unsigned l = lds_base + row * LDS_ROW + sub * 16;
    asm volatile("global_load_async_to_lds_b128 %0, %1, off" ::"v"(l), "v"(g) : "memory");
  }
}

// ---------------- generic fused WMMA GEMM (64x64 block tile, async dbl-buffer) --
struct GemmP {
  const _Float16* A;
  const _Float16* B;   // pre-transposed weight: [K][N]
  const float* bias;   // may be null
  const float* r0;     // residual 0 (f32, ldo layout) or null
  const float* r1;     // residual 1 or null
  float* oF;
  _Float16* oH;
  int M, N, K, lda, ldb, ldo;
  float scale;
  int act;    // 0 none, 1 relu, 2 exact gelu
  int okind;  // 0 f32 row-major, 1 f16 row-major, 2 q/v scatter [b,h,n,d],
              // 3 kT scatter [b,h,d,n], 4 f32 row-major + sinusoid pos-embed
};

__global__ void __launch_bounds__(128) k_gemm(GemmP p) {
  __shared__ __align__(16) char bsl[2][LDS_SLAB];
  int tid = threadIdx.x;
  int lane = tid & 31, wv = tid >> 5;
  int nb = p.N >> 6;
  int m0 = (blockIdx.x / nb) * 64 + wv * 16;
  int n0 = (blockIdx.x % nb) * 64;
  unsigned lds0 = (unsigned)(size_t)&bsl[0][0];  // low 32 bits == LDS offset

  // strength-reduced per-lane A pointer (row lane&15, K-half by lane>>4)
  const _Float16* aptr = p.A + (size_t)(m0 + (lane & 15)) * p.lda + 8 * (lane >> 4);

  v8f acc[4] = {};
  int KT = p.K >> 5;
  slab_fill_async(p.B, p.ldb, n0, 0, lds0, tid);  // prologue: fill buffer 0
  for (int kt = 0; kt < KT; ++kt) {
    int cur = kt & 1;
    asm volatile("s_wait_asynccnt 0x0" ::: "memory");  // my slab writes landed
    __syncthreads();                                   // everyone's writes landed
    if (kt + 1 < KT)  // overlap: fill next slab while computing on cur
      slab_fill_async(p.B, p.ldb, n0, (kt + 1) << 5, lds0 + (cur ^ 1) * LDS_SLAB, tid);
    v16h a;
    {
      union { v16h v; uint4 q[2]; } u;
      u.q[0] = *(const uint4*)(aptr);
      u.q[1] = *(const uint4*)(aptr + 16);
      a = u.v;
    }
    if (kt + 2 < KT) __builtin_prefetch(aptr + 64, 0, 1);
    aptr += 32;
    const char* pb = &bsl[cur][0];
    v16h b0 = ld_b_lds(pb, lane, 0);
    v16h b1 = ld_b_lds(pb, lane, 1);
    v16h b2 = ld_b_lds(pb, lane, 2);
    v16h b3 = ld_b_lds(pb, lane, 3);
    SCHED_FENCE();  // issue all B-fragment LDS loads (4 x 16 16-bit) first, then the 4 WMMAs back-to-back
    acc[0] = WMMA16(a, b0, acc[0]);
    acc[1] = WMMA16(a, b1, acc[1]);
    acc[2] = WMMA16(a, b2, acc[2]);
    acc[3] = WMMA16(a, b3, acc[3]);
  }

  int col = lane & 15, rb = (lane >> 4) * 8;
#pragma unroll
  for (int j = 0; j < 4; ++j) {
#pragma unroll
    for (int v = 0; v < 8; ++v) {
      int mm = m0 + rb + v;
      int nn = n0 + 16 * j + col;
      float val = acc[j][v];
      if (p.bias) val += p.bias[nn];
      val *= p.scale;
      if (p.act == 1) val = fmaxf(val, 0.0f);
      else if (p.act == 2) val = 0.5f * val * (1.0f + erff(val * 0.70710678118f));
      if (p.okind == 0 || p.okind == 4) {
        if (p.okind == 4) {  // fixed sinusoid positional embedding
          int n = mm % NTOK;
          float ex = (float)((nn >> 1) << 1) * (1.0f / (float)CDIM);
          float ang = (float)n * __expf(-9.210340371976184f * ex);
          val += (nn & 1) ? __cosf(ang) : __sinf(ang);
        }
        size_t oi = (size_t)mm * p.ldo + nn;
        if (p.r0) val += p.r0[oi];
        if (p.r1) val += p.r1[oi];
        p.oF[oi] = val;
      } else if (p.okind == 1) {
        p.oH[(size_t)mm * p.ldo + nn] = (_Float16)val;
      } else {  // per-head scatter
        int b = mm / NTOK, n = mm % NTOK;
        int h = nn >> 6, d = nn & 63;
        size_t bh = (size_t)(b * NHEAD + h);
        if (p.okind == 2) p.oH[(bh * NTOK + n) * HDIM + d] = (_Float16)val;
        else              p.oH[(bh * HDIM + d) * NTOK + n] = (_Float16)val;
      }
    }
  }
}

// ---------------- weight convert + transpose: WT[k][o] = W[o][k] --------------
__global__ void __launch_bounds__(256) k_wt(const float* __restrict__ W,
                                            _Float16* __restrict__ WT, int O, int K) {
  size_t i = (size_t)blockIdx.x * blockDim.x + threadIdx.x;
  if (i >= (size_t)O * K) return;
  int k = (int)(i % K);
  int o = (int)(i / K);
  WT[(size_t)k * O + o] = (_Float16)W[i];
}

// ---------------- im2col for Conv3d(stride==kernel) ---------------------------
__global__ void __launch_bounds__(256) k_im2col(const float* __restrict__ x,
                                                _Float16* __restrict__ P) {
  size_t i = (size_t)blockIdx.x * blockDim.x + threadIdx.x;
  if (i >= (size_t)MROWS * KCONV) return;
  int k = (int)(i % KCONV);
  int m = (int)(i / KCONV);
  int b = m / NTOK, s = m % NTOK;
  int t = s / 196, r = s % 196;
  int hh = r / 14, ww = r % 14;
  int ci = k / 512, r2 = k % 512;
  int kt = r2 / 256, kh = (r2 % 256) / 16, kw = r2 % 16;
  size_t xi = ((((size_t)b * 3 + ci) * 16 + (2 * t + kt)) * 224 + (16 * hh + kh)) * 224 +
              (16 * ww + kw);
  P[i] = (_Float16)x[xi];
}

// ---------------- LayerNorm (one row per workgroup) ---------------------------
__global__ void __launch_bounds__(256) k_layernorm(const float* __restrict__ X,
                                                   const float* __restrict__ g,
                                                   const float* __restrict__ b,
                                                   _Float16* outH, float* outF, int C) {
  int row = blockIdx.x;
  const float* xr = X + (size_t)row * C;
  int tid = threadIdx.x, lane = tid & 31, wid = tid >> 5;
  __shared__ float swave[8];
  __shared__ float sbc[2];
  float s = 0.0f;
  for (int c = tid; c < C; c += blockDim.x) s += xr[c];
#pragma unroll
  for (int off = 16; off > 0; off >>= 1) s += __shfl_down(s, off, 32);
  if (lane == 0) swave[wid] = s;
  __syncthreads();
  if (tid == 0) {
    float t = 0.0f;
    for (int i = 0; i < 8; ++i) t += swave[i];
    sbc[0] = t / (float)C;
  }
  __syncthreads();
  float mean = sbc[0];
  float vs = 0.0f;
  for (int c = tid; c < C; c += blockDim.x) {
    float d = xr[c] - mean;
    vs += d * d;
  }
#pragma unroll
  for (int off = 16; off > 0; off >>= 1) vs += __shfl_down(vs, off, 32);
  __syncthreads();
  if (lane == 0) swave[wid] = vs;
  __syncthreads();
  if (tid == 0) {
    float t = 0.0f;
    for (int i = 0; i < 8; ++i) t += swave[i];
    sbc[1] = rsqrtf(t / (float)C + 1e-5f);
  }
  __syncthreads();
  float rs = sbc[1];
  for (int c = tid; c < C; c += blockDim.x) {
    float val = (xr[c] - mean) * rs * g[c] + b[c];
    if (outH) outH[(size_t)row * C + c] = (_Float16)val;
    else      outF[(size_t)row * C + c] = val;
  }
}

// ---------------- flash attention: 1 wave = 16 queries of one (b,h) -----------
__global__ void __launch_bounds__(128) k_attn(const _Float16* __restrict__ Q,
                                              const _Float16* __restrict__ KT,
                                              const _Float16* __restrict__ V,
                                              _Float16* __restrict__ O) {
  __shared__ __align__(16) _Float16 pbuf[4][16 * 40];  // P tile staging, padded rows
  int lane = threadIdx.x & 31;
  int ws = threadIdx.x >> 5;
  int gw = blockIdx.x * 4 + ws;                 // grid sized exactly: no tail
  int qt = gw % (NTOK / 16);
  int bh = gw / (NTOK / 16);
  const _Float16* Qb = Q + (size_t)bh * NTOK * HDIM;   // [n][d], q pre-scaled
  const _Float16* Kb = KT + (size_t)bh * HDIM * NTOK;  // [d][n]
  const _Float16* Vb = V + (size_t)bh * NTOK * HDIM;   // [n][d]
  _Float16* pb = &pbuf[ws][0];
  v16h qa0 = ld_a_frag(Qb, HDIM, qt * 16, 0, lane);
  v16h qa1 = ld_a_frag(Qb, HDIM, qt * 16, 32, lane);
  v8f oacc[4] = {};
  float mi[8], li[8];
#pragma unroll
  for (int v = 0; v < 8; ++v) { mi[v] = -3.0e30f; li[v] = 0.0f; }
  int col = lane & 15, half = lane >> 4;
  for (int kt = 0; kt < NTOK / 32; ++kt) {
    v8f s0 = {}, s1 = {};
    v16h b00 = ld_b_frag(Kb, NTOK, 0, kt * 32, lane);
    v16h b01 = ld_b_frag(Kb, NTOK, 0, kt * 32 + 16, lane);
    v16h b10 = ld_b_frag(Kb, NTOK, 32, kt * 32, lane);
    v16h b11 = ld_b_frag(Kb, NTOK, 32, kt * 32 + 16, lane);
    SCHED_FENCE();  // all 4 K-fragment loads issued before the WMMA chain
    s0 = WMMA16(qa0, b00, s0);
    s1 = WMMA16(qa0, b01, s1);
    s0 = WMMA16(qa1, b10, s0);
    s1 = WMMA16(qa1, b11, s1);
    // online softmax: row-wise max over 32 key cols (16 col-lanes per half)
    float mn[8];
#pragma unroll
    for (int v = 0; v < 8; ++v) mn[v] = fmaxf(s0[v], s1[v]);
#pragma unroll
    for (int off = 1; off < 16; off <<= 1)
#pragma unroll
      for (int v = 0; v < 8; ++v) mn[v] = fmaxf(mn[v], __shfl_xor(mn[v], off, 32));
    float p0[8], p1[8], rsum[8];
#pragma unroll
    for (int v = 0; v < 8; ++v) {
      float m2 = fmaxf(mi[v], mn[v]);
      float f = __expf(mi[v] - m2);
      mi[v] = m2;
      li[v] *= f;
      oacc[0][v] *= f; oacc[1][v] *= f; oacc[2][v] *= f; oacc[3][v] *= f;
      p0[v] = __expf(s0[v] - m2);
      p1[v] = __expf(s1[v] - m2);
      rsum[v] = p0[v] + p1[v];
    }
#pragma unroll
    for (int off = 1; off < 16; off <<= 1)
#pragma unroll
      for (int v = 0; v < 8; ++v) rsum[v] += __shfl_xor(rsum[v], off, 32);
#pragma unroll
    for (int v = 0; v < 8; ++v) li[v] += rsum[v];
    // re-layout P (C-frag -> A-frag) through LDS
#pragma unroll
    for (int v = 0; v < 8; ++v) {
      int r = v + 8 * half;
      pb[r * 40 + col] = (_Float16)p0[v];
      pb[r * 40 + 16 + col] = (_Float16)p1[v];
    }
    __syncthreads();
    v16h pa;
    {
      union { v16h v; uint4 q[2]; } u;
      int r = lane & 15, h2 = lane >> 4;
      u.q[0] = *(const uint4*)&pb[r * 40 + 8 * h2];
      u.q[1] = *(const uint4*)&pb[r * 40 + 16 + 8 * h2];
      pa = u.v;
    }
    __syncthreads();
    v16h bv0 = ld_b_frag(Vb, HDIM, kt * 32, 0, lane);
    v16h bv1 = ld_b_frag(Vb, HDIM, kt * 32, 16, lane);
    v16h bv2 = ld_b_frag(Vb, HDIM, kt * 32, 32, lane);
    v16h bv3 = ld_b_frag(Vb, HDIM, kt * 32, 48, lane);
    SCHED_FENCE();  // all 4 V-fragment loads issued before the WMMA chain
    oacc[0] = WMMA16(pa, bv0, oacc[0]);
    oacc[1] = WMMA16(pa, bv1, oacc[1]);
    oacc[2] = WMMA16(pa, bv2, oacc[2]);
    oacc[3] = WMMA16(pa, bv3, oacc[3]);
  }
  int b = bh / NHEAD, h = bh % NHEAD;
  float inv[8];
#pragma unroll
  for (int v = 0; v < 8; ++v) inv[v] = 1.0f / li[v];
#pragma unroll
  for (int nc = 0; nc < 4; ++nc)
#pragma unroll
    for (int v = 0; v < 8; ++v) {
      int r = v + 8 * half;
      O[((size_t)b * NTOK + qt * 16 + r) * CDIM + h * HDIM + nc * 16 + col] =
          (_Float16)(oacc[nc][v] * inv[v]);
    }
}

// ---------------- host-side orchestration -------------------------------------
static inline void gemm(hipStream_t st, const _Float16* A, int lda, const _Float16* Bm,
                        int M, int N, int K, const float* bias, float scale, int act,
                        const float* r0, const float* r1, float* oF, _Float16* oH,
                        int ldo, int okind) {
  GemmP p;
  p.A = A; p.B = Bm; p.bias = bias; p.r0 = r0; p.r1 = r1; p.oF = oF; p.oH = oH;
  p.M = M; p.N = N; p.K = K; p.lda = lda; p.ldb = N; p.ldo = ldo;
  p.scale = scale; p.act = act; p.okind = okind;
  int blocks = (M / 64) * (N / 64);  // exact: M=3136=49*64, N in {64,768,3072}
  k_gemm<<<blocks, 128, 0, st>>>(p);
}

static inline void wtx(hipStream_t st, const float* W, _Float16* WT, int O, int K) {
  size_t n = (size_t)O * K;
  k_wt<<<(int)((n + 255) / 256), 256, 0, st>>>(W, WT, O, K);
}

extern "C" void kernel_launch(void* const* d_in, const int* in_sizes, int n_in,
                              void* d_out, int out_size, void* d_ws, size_t ws_size,
                              hipStream_t stream) {
  const float* x      = (const float*)d_in[0];
  const float* conv_w = (const float*)d_in[1];
  const float* conv_b = (const float*)d_in[2];
  const float* ln1_g  = (const float*)d_in[3];
  const float* ln1_b  = (const float*)d_in[4];
  const float* q_w    = (const float*)d_in[5];
  const float* q_b    = (const float*)d_in[6];
  const float* k_w    = (const float*)d_in[7];
  const float* v_w    = (const float*)d_in[8];
  const float* v_b    = (const float*)d_in[9];
  const float* proj_w = (const float*)d_in[10];
  const float* proj_b = (const float*)d_in[11];
  const float* ln2_g  = (const float*)d_in[12];
  const float* ln2_b  = (const float*)d_in[13];
  const float* fc1_w  = (const float*)d_in[14];
  const float* fc1_b  = (const float*)d_in[15];
  const float* fc2_w  = (const float*)d_in[16];
  const float* fc2_b  = (const float*)d_in[17];
  const float* ada_g  = (const float*)d_in[18];
  const float* ada_b  = (const float*)d_in[19];
  const float* down_w = (const float*)d_in[20];
  const float* down_b = (const float*)d_in[21];
  const float* up_w   = (const float*)d_in[22];
  const float* up_b   = (const float*)d_in[23];
  const float* normf_g = (const float*)d_in[24];
  const float* normf_b = (const float*)d_in[25];

  char* wp = (char*)d_ws;
  auto take = [&](size_t bytes) -> void* {
    void* p = (void*)wp;
    wp += (bytes + 255) & ~(size_t)255;
    return p;
  };
  _Float16* wT  = (_Float16*)take((size_t)2359296 * 2);          // reused weight slab
  _Float16* Pim = (_Float16*)take((size_t)MROWS * KCONV * 2);    // im2col
  float*    tok = (float*)take((size_t)MROWS * CDIM * 4);
  _Float16* hb  = (_Float16*)take((size_t)MROWS * CDIM * 2);     // LN output (f16)
  _Float16* ffb = (_Float16*)take((size_t)MROWS * FFDIM * 2);
  _Float16* dwb = (_Float16*)take((size_t)MROWS * BOTD * 2);
  _Float16* qb  = (_Float16*)take((size_t)MROWS * CDIM * 2);
  _Float16* ktb = (_Float16*)take((size_t)MROWS * CDIM * 2);
  _Float16* vb  = (_Float16*)take((size_t)MROWS * CDIM * 2);
  _Float16* ob  = (_Float16*)take((size_t)MROWS * CDIM * 2);
  float*    adp = (float*)take((size_t)MROWS * CDIM * 4);

  // ---- patch embed: im2col GEMM + bias + sinusoid pos-embed -> tok (f32)
  wtx(stream, conv_w, wT, CDIM, KCONV);
  {
    size_t n = (size_t)MROWS * KCONV;
    k_im2col<<<(int)((n + 255) / 256), 256, 0, stream>>>(x, Pim);
  }
  gemm(stream, Pim, KCONV, wT, MROWS, CDIM, KCONV, conv_b, 1.0f, 0, nullptr, nullptr,
       tok, nullptr, CDIM, 4);

  const float qscale = 0.125f;  // HD^-0.5
  for (int i = 0; i < DEPTH; ++i) {
    // attention branch
    k_layernorm<<<MROWS, 256, 0, stream>>>(tok, ln1_g + i * CDIM, ln1_b + i * CDIM,
                                           hb, nullptr, CDIM);
    wtx(stream, q_w + (size_t)i * CDIM * CDIM, wT, CDIM, CDIM);
    gemm(stream, hb, CDIM, wT, MROWS, CDIM, CDIM, q_b + i * CDIM, qscale, 0,
         nullptr, nullptr, nullptr, qb, CDIM, 2);
    wtx(stream, k_w + (size_t)i * CDIM * CDIM, wT, CDIM, CDIM);
    gemm(stream, hb, CDIM, wT, MROWS, CDIM, CDIM, nullptr, 1.0f, 0,
         nullptr, nullptr, nullptr, ktb, CDIM, 3);
    wtx(stream, v_w + (size_t)i * CDIM * CDIM, wT, CDIM, CDIM);
    gemm(stream, hb, CDIM, wT, MROWS, CDIM, CDIM, v_b + i * CDIM, 1.0f, 0,
         nullptr, nullptr, nullptr, vb, CDIM, 2);
    k_attn<<<(BATCH * NHEAD * (NTOK / 16)) / 4, 128, 0, stream>>>(qb, ktb, vb, ob);
    wtx(stream, proj_w + (size_t)i * CDIM * CDIM, wT, CDIM, CDIM);
    gemm(stream, ob, CDIM, wT, MROWS, CDIM, CDIM, proj_b + i * CDIM, 1.0f, 0,
         tok, nullptr, tok, nullptr, CDIM, 0);
    // parallel adapter: LN -> down -> ReLU -> up -> *0.1
    k_layernorm<<<MROWS, 256, 0, stream>>>(tok, ada_g + i * CDIM, ada_b + i * CDIM,
                                           hb, nullptr, CDIM);
    wtx(stream, down_w + (size_t)i * BOTD * CDIM, wT, BOTD, CDIM);
    gemm(stream, hb, CDIM, wT, MROWS, BOTD, CDIM, down_b + i * BOTD, 1.0f, 1,
         nullptr, nullptr, nullptr, dwb, BOTD, 1);
    wtx(stream, up_w + (size_t)i * CDIM * BOTD, wT, CDIM, BOTD);
    gemm(stream, dwb, BOTD, wT, MROWS, CDIM, BOTD, up_b + i * CDIM, 0.1f, 0,
         nullptr, nullptr, adp, nullptr, CDIM, 0);
    // MLP branch, fused residual: tok = tok + mlp + adapt
    k_layernorm<<<MROWS, 256, 0, stream>>>(tok, ln2_g + i * CDIM, ln2_b + i * CDIM,
                                           hb, nullptr, CDIM);
    wtx(stream, fc1_w + (size_t)i * FFDIM * CDIM, wT, FFDIM, CDIM);
    gemm(stream, hb, CDIM, wT, MROWS, FFDIM, CDIM, fc1_b + i * FFDIM, 1.0f, 2,
         nullptr, nullptr, nullptr, ffb, FFDIM, 1);
    wtx(stream, fc2_w + (size_t)i * CDIM * FFDIM, wT, CDIM, FFDIM);
    gemm(stream, ffb, FFDIM, wT, MROWS, CDIM, FFDIM, fc2_b + i * CDIM, 1.0f, 0,
         tok, adp, tok, nullptr, CDIM, 0);
  }
  // final layer norm -> f32 output
  k_layernorm<<<MROWS, 256, 0, stream>>>(tok, normf_g, normf_b, nullptr,
                                         (float*)d_out, CDIM);
}
